// PointcloudTokenizer_38577396252864
// MI455X (gfx1250) — matrix operand order repaired and checked by reading, not compiled, not run on hardware
//
#include <hip/hip_runtime.h>

// ---------------------------------------------------------------------------
// Problem constants (match reference)
// ---------------------------------------------------------------------------
static constexpr int Bn   = 8;
static constexpr int Nn   = 16384;
static constexpr int Gn   = 512;
static constexpr int Sn   = 32;
static constexpr int TDn  = 384;
static constexpr int NGRP = Bn * Gn;          // 4096 groups
static constexpr float INV_CNT = 1.0f / float(NGRP * Sn); // 1/131072

typedef __attribute__((ext_vector_type(16))) _Float16 v16h;
typedef __attribute__((ext_vector_type(8)))  float    v8f;

// ---------------------------------------------------------------------------
// WMMA helpers (CDNA5 wave32, D = A(16x32 f16) * B(32x16 f16) + C(f32))
// ---------------------------------------------------------------------------
__device__ __forceinline__ v8f wmma_f16(v16h a, v16h b, v8f c) {
  return __builtin_amdgcn_wmma_f32_16x16x32_f16(false, a, false, b, (short)0, c,
                                                false, false);
}

// A fragment: 16x32 f16 tile from row-major [M x K] buffer.
// Lane L<16: M=L, K = kb..kb+15 with kb=k0; L>=16: kb=k0+16.
__device__ __forceinline__ v16h load_frag_a(const _Float16* __restrict__ base,
                                            int row0, int k0, int ldk) {
  int lane = threadIdx.x & 31;
  int m    = row0 + (lane & 15);
  int kb   = k0 + ((lane >> 4) << 4);
  const _Float16* p = base + (size_t)m * ldk + kb;
  v16h f;
#pragma unroll
  for (int e = 0; e < 8; ++e)  f[e] = p[e];
#pragma unroll
  for (int e = 8; e < 16; ++e) f[e] = p[e];   // K = kb + e
  return f;
}

// B fragment: 32x16 f16 tile; weights stored row-major [N x K] (w[n][k]),
// so B[k][n] = w[n][k]. Lane L: n=n0+(L&15), K = k0 + (L>=16?16:0) + e.
// 16 consecutive halves -> single 32B vector load.
__device__ __forceinline__ v16h load_frag_b(const _Float16* __restrict__ w,
                                            int n0, int k0, int ldk) {
  int lane = threadIdx.x & 31;
  int n    = n0 + (lane & 15);
  int kb   = k0 + ((lane >> 4) << 4);
  return *(const v16h*)(w + (size_t)n * ldk + kb);
}

// ---------------------------------------------------------------------------
// Kernel 1: farthest point sampling. One block per batch, 512 threads,
// 32 points per thread kept in registers.
// ---------------------------------------------------------------------------
__global__ __launch_bounds__(512) void fps_kernel(
    const float* __restrict__ pts, float* __restrict__ centers) {
  const int b   = blockIdx.x;
  const int tid = threadIdx.x;
  const float* __restrict__ P = pts + (size_t)b * Nn * 3;

  __shared__ float rv[16];
  __shared__ int   ri[16];
  __shared__ int   bcast;

  float mind[32];
#pragma unroll
  for (int j = 0; j < 32; ++j) mind[j] = 1e10f;

  float lx = P[0], ly = P[1], lz = P[2];
  if (tid == 0) {
    float* c0 = centers + (size_t)(b * Gn) * 3;
    c0[0] = lx; c0[1] = ly; c0[2] = lz;
  }

  for (int i = 1; i < Gn; ++i) {
    float bv = -1.0f; int bp = 0;
#pragma unroll
    for (int j = 0; j < 32; ++j) {
      int p = tid + (j << 9);
      float dx = P[p * 3 + 0] - lx;
      float dy = P[p * 3 + 1] - ly;
      float dz = P[p * 3 + 2] - lz;
      float d  = dx * dx + dy * dy + dz * dz;
      float mj = fminf(mind[j], d);
      mind[j]  = mj;
      if (mj > bv) { bv = mj; bp = p; }
    }
    // wave32 argmax
#pragma unroll
    for (int off = 16; off >= 1; off >>= 1) {
      float ov = __shfl_down(bv, off);
      int   op = __shfl_down(bp, off);
      if (ov > bv || (ov == bv && op < bp)) { bv = ov; bp = op; }
    }
    if ((tid & 31) == 0) { rv[tid >> 5] = bv; ri[tid >> 5] = bp; }
    __syncthreads();
    if (tid < 32) {
      float v2 = (tid < 16) ? rv[tid] : -2.0f;
      int   p2 = (tid < 16) ? ri[tid] : 0;
#pragma unroll
      for (int off = 8; off >= 1; off >>= 1) {
        float ov = __shfl_down(v2, off);
        int   op = __shfl_down(p2, off);
        if (ov > v2 || (ov == v2 && op < p2)) { v2 = ov; p2 = op; }
      }
      if (tid == 0) bcast = p2;
    }
    __syncthreads();
    const int w = bcast;
    lx = P[w * 3 + 0]; ly = P[w * 3 + 1]; lz = P[w * 3 + 2];
    if (tid == 0) {
      float* cp = centers + ((size_t)b * Gn + i) * 3;
      cp[0] = lx; cp[1] = ly; cp[2] = lz;
    }
    __syncthreads();
  }
}

// ---------------------------------------------------------------------------
// Kernel 2: KNN (top-32 nearest) + gather + center-subtract.
// One block per (b,g) center, 256 threads, 64 candidate distances in regs.
// ---------------------------------------------------------------------------
__global__ __launch_bounds__(256) void knn_kernel(
    const float* __restrict__ pts, const float* __restrict__ centers,
    float* __restrict__ grp) {
  const int gi  = blockIdx.x;
  const int b   = gi >> 9;
  const int tid = threadIdx.x;
  const float* __restrict__ P = pts + (size_t)b * Nn * 3;

  const float cx = centers[(size_t)gi * 3 + 0];
  const float cy = centers[(size_t)gi * 3 + 1];
  const float cz = centers[(size_t)gi * 3 + 2];

  float dl[64];
#pragma unroll
  for (int j = 0; j < 64; ++j) {
    int p = tid + (j << 8);
    float dx = P[p * 3 + 0] - cx;
    float dy = P[p * 3 + 1] - cy;
    float dz = P[p * 3 + 2] - cz;
    dl[j] = dx * dx + dy * dy + dz * dz;
  }

  __shared__ float rv[8];
  __shared__ int   ri[8];
  __shared__ int   bc;

  for (int r = 0; r < Sn; ++r) {
    float bv = 1e30f; int bj = 0;
#pragma unroll
    for (int j = 0; j < 64; ++j)
      if (dl[j] < bv) { bv = dl[j]; bj = j; }
    int bp = tid + (bj << 8);
#pragma unroll
    for (int off = 16; off >= 1; off >>= 1) {
      float ov = __shfl_down(bv, off);
      int   op = __shfl_down(bp, off);
      if (ov < bv || (ov == bv && op < bp)) { bv = ov; bp = op; }
    }
    if ((tid & 31) == 0) { rv[tid >> 5] = bv; ri[tid >> 5] = bp; }
    __syncthreads();
    if (tid < 32) {
      float v2 = (tid < 8) ? rv[tid] : 1e31f;
      int   p2 = (tid < 8) ? ri[tid] : 0;
#pragma unroll
      for (int off = 4; off >= 1; off >>= 1) {
        float ov = __shfl_down(v2, off);
        int   op = __shfl_down(p2, off);
        if (ov < v2 || (ov == v2 && op < p2)) { v2 = ov; p2 = op; }
      }
      if (tid == 0) bc = p2;
    }
    __syncthreads();
    const int wp = bc;
    // mark selected (static indexing only -> stays in registers)
#pragma unroll
    for (int j = 0; j < 64; ++j)
      if ((tid + (j << 8)) == wp) dl[j] = 1e30f;
    if (tid == r) {
      float* o = grp + ((size_t)gi * Sn + r) * 3;
      o[0] = P[wp * 3 + 0] - cx;
      o[1] = P[wp * 3 + 1] - cy;
      o[2] = P[wp * 3 + 2] - cz;
    }
    __syncthreads();
  }
}

// ---------------------------------------------------------------------------
// Small utility kernels
// ---------------------------------------------------------------------------
__global__ void zero_kernel(float* __restrict__ p, int n) {
  int i = blockIdx.x * blockDim.x + threadIdx.x;
  if (i < n) p[i] = 0.0f;
}

__global__ void f32_to_f16_kernel(const float* __restrict__ src,
                                  _Float16* __restrict__ dst, int n) {
  int i = blockIdx.x * blockDim.x + threadIdx.x;
  if (i < n) dst[i] = (_Float16)src[i];
}

// BN1 raw statistics: one block per group, 128 threads (one per channel).
__global__ __launch_bounds__(128) void stats1_kernel(
    const float* __restrict__ grp, const float* __restrict__ w1,
    float* __restrict__ sums1, float* __restrict__ sumsq1) {
  const int gi = blockIdx.x;
  const int c  = threadIdx.x;
  __shared__ float Xs[Sn * 3];
  if (c < Sn * 3) Xs[c] = grp[(size_t)gi * Sn * 3 + c];
  __syncthreads();
  const float w0 = w1[c * 3 + 0], wa = w1[c * 3 + 1], wb = w1[c * 3 + 2];
  float s = 0.0f, q = 0.0f;
#pragma unroll
  for (int p = 0; p < Sn; ++p) {
    float v = Xs[p * 3 + 0] * w0 + Xs[p * 3 + 1] * wa + Xs[p * 3 + 2] * wb;
    s += v; q += v * v;
  }
  atomicAdd(&sums1[c], s);
  atomicAdd(&sumsq1[c], q);
}

// mean/var -> affine scale/shift:  y = x*scale + shift
__global__ void bn_finalize_kernel(const float* __restrict__ sums,
                                   const float* __restrict__ sumsq,
                                   const float* __restrict__ g,
                                   const float* __restrict__ be,
                                   float* __restrict__ scale,
                                   float* __restrict__ shift, int nch) {
  int c = blockIdx.x * blockDim.x + threadIdx.x;
  if (c >= nch) return;
  float mean = sums[c] * INV_CNT;
  float var  = sumsq[c] * INV_CNT - mean * mean;
  float sc   = g[c] * rsqrtf(var + 1e-5f);
  scale[c] = sc;
  shift[c] = be[c] - mean * sc;
}

// ---------------------------------------------------------------------------
// Mini-PointNet, one block (8 waves) per group. FINAL=0: accumulate BN3
// raw stats of f3. FINAL=1: full pipeline to tokens.
// LDS (64KB):  [0,8K) f1h(32x128 f16) | [8K,8.5K) Xs | [32K,64K) hbuf(32x512)
//              [0,32K) h3 (phase D input, reuses dead f1h/Xs)
// ---------------------------------------------------------------------------
template <int FINAL>
__global__ __launch_bounds__(256) void pointnet_kernel(
    const float* __restrict__ grp,
    const float* __restrict__ w1,
    const float* __restrict__ scale1, const float* __restrict__ shift1,
    const _Float16* __restrict__ w2h, const float* __restrict__ b2,
    const _Float16* __restrict__ w3h,
    float* __restrict__ sums3, float* __restrict__ sumsq3,
    const float* __restrict__ scale3, const float* __restrict__ shift3,
    const _Float16* __restrict__ w4h, const float* __restrict__ b4,
    float* __restrict__ tokens, float* __restrict__ mask) {
  __shared__ __align__(16) unsigned char smem[65536];
  _Float16* f1h  = (_Float16*)(smem);            // 32x128
  float*    Xs   = (float*)(smem + 8192);        // 32x3
  _Float16* hbuf = (_Float16*)(smem + 32768);    // 32x512 concat [fg | f2]
  _Float16* h3   = (_Float16*)(smem);            // 32x512 relu(bn3(f3))

  const int gi   = blockIdx.x;
  const int tid  = threadIdx.x;
  const int wave = tid >> 5;
  const int lane = tid & 31;

  if (tid < Sn * 3) Xs[tid] = grp[(size_t)gi * Sn * 3 + tid];
  // warm L2 path for the big weight matrix
  __builtin_prefetch(w3h + (size_t)tid * 64, 0, 1);
  __syncthreads();

  // ---- Phase A: f1 = relu(bn1(X @ w1^T)) -> f16 LDS ----------------------
  for (int idx = tid; idx < Sn * 128; idx += 256) {
    int s = idx >> 7, c = idx & 127;
    float v = Xs[s * 3 + 0] * w1[c * 3 + 0] +
              Xs[s * 3 + 1] * w1[c * 3 + 1] +
              Xs[s * 3 + 2] * w1[c * 3 + 2];
    v = fmaxf(v * scale1[c] + shift1[c], 0.0f);
    f1h[s * 128 + c] = (_Float16)v;
  }
  __syncthreads();

  // ---- Phase B: f2 = f1 @ w2^T + b2   (M=32, N=256, K=128) ---------------
  for (int t = 0; t < 2; ++t) {
    const int n0 = (wave * 2 + t) * 16;
    v8f acc0 = {}; v8f acc1 = {};
    for (int kk = 0; kk < 128; kk += 32) {
      v16h bf = load_frag_b(w2h, n0, kk, 128);
      v16h a0 = load_frag_a(f1h, 0, kk, 128);
      v16h a1 = load_frag_a(f1h, 16, kk, 128);
      acc0 = wmma_f16(a0, bf, acc0);
      acc1 = wmma_f16(a1, bf, acc1);
    }
    const int   n     = n0 + (lane & 15);
    const float bias  = b2[n];
    const int   rbase = (lane >> 4) << 3;
#pragma unroll
    for (int r = 0; r < 8; ++r) {
      hbuf[(r + rbase) * 512 + 256 + n]      = (_Float16)(acc0[r] + bias);
      hbuf[(16 + r + rbase) * 512 + 256 + n] = (_Float16)(acc1[r] + bias);
    }
  }
  __syncthreads();

  // ---- global-feature column max -> broadcast into hbuf[:,0:256] ---------
  {
    const int c = tid;
    float m = -1e30f;
#pragma unroll
    for (int s = 0; s < Sn; ++s) m = fmaxf(m, (float)hbuf[s * 512 + 256 + c]);
    const _Float16 mh = (_Float16)m;
#pragma unroll
    for (int s = 0; s < Sn; ++s) hbuf[s * 512 + c] = mh;
  }
  __syncthreads();

  // ---- Phase C: f3 = h @ w3^T   (M=32, N=512, K=512) ---------------------
  for (int t = 0; t < 4; ++t) {
    const int n0 = (wave * 4 + t) * 16;
    v8f acc0 = {}; v8f acc1 = {};
    for (int kk = 0; kk < 512; kk += 32) {
      v16h bf = load_frag_b(w3h, n0, kk, 512);
      v16h a0 = load_frag_a(hbuf, 0, kk, 512);
      v16h a1 = load_frag_a(hbuf, 16, kk, 512);
      acc0 = wmma_f16(a0, bf, acc0);
      acc1 = wmma_f16(a1, bf, acc1);
    }
    const int n     = n0 + (lane & 15);
    const int rbase = (lane >> 4) << 3;
    if (FINAL == 0) {
      float s = 0.0f, q = 0.0f;
#pragma unroll
      for (int r = 0; r < 8; ++r) {
        s += acc0[r] + acc1[r];
        q += acc0[r] * acc0[r] + acc1[r] * acc1[r];
      }
      s += __shfl_xor(s, 16);
      q += __shfl_xor(q, 16);
      if (lane < 16) {
        atomicAdd(&sums3[n], s);
        atomicAdd(&sumsq3[n], q);
      }
    } else {
      const float sc = scale3[n], sh = shift3[n];
#pragma unroll
      for (int r = 0; r < 8; ++r) {
        h3[(r + rbase) * 512 + n]      = (_Float16)fmaxf(acc0[r] * sc + sh, 0.0f);
        h3[(16 + r + rbase) * 512 + n] = (_Float16)fmaxf(acc1[r] * sc + sh, 0.0f);
      }
    }
  }
  if (FINAL == 0) return;
  __syncthreads();

  // ---- Phase D: tokens = colmax(f4 = h3 @ w4^T + b4)   (N=384, K=512) ----
  for (int t = 0; t < 3; ++t) {
    const int n0 = (wave * 3 + t) * 16;
    v8f acc0 = {}; v8f acc1 = {};
    for (int kk = 0; kk < 512; kk += 32) {
      v16h bf = load_frag_b(w4h, n0, kk, 512);
      v16h a0 = load_frag_a(h3, 0, kk, 512);
      v16h a1 = load_frag_a(h3, 16, kk, 512);
      acc0 = wmma_f16(a0, bf, acc0);
      acc1 = wmma_f16(a1, bf, acc1);
    }
    const int n = n0 + (lane & 15);
    float m = -1e30f;
#pragma unroll
    for (int r = 0; r < 8; ++r) m = fmaxf(m, fmaxf(acc0[r], acc1[r]));
    m = fmaxf(m, __shfl_xor(m, 16));   // combine rows {.. } with {..+8/+24}
    if (lane < 16) tokens[(size_t)gi * TDn + n] = m + b4[n];
  }
  if (tid == 0) mask[gi] = 1.0f;
}

// ---------------------------------------------------------------------------
// Host launcher
// ---------------------------------------------------------------------------
extern "C" void kernel_launch(void* const* d_in, const int* in_sizes, int n_in,
                              void* d_out, int out_size, void* d_ws,
                              size_t ws_size, hipStream_t stream) {
  const float* points = (const float*)d_in[0];
  const float* w1 = (const float*)d_in[2];
  const float* g1 = (const float*)d_in[3];
  const float* be1 = (const float*)d_in[4];
  const float* w2 = (const float*)d_in[5];
  const float* b2 = (const float*)d_in[6];
  const float* w3 = (const float*)d_in[7];
  const float* g3 = (const float*)d_in[8];
  const float* be3 = (const float*)d_in[9];
  const float* w4 = (const float*)d_in[10];
  const float* b4 = (const float*)d_in[11];

  float* out     = (float*)d_out;
  float* tokens  = out;                          // B*G*TD
  float* centers = out + (size_t)NGRP * TDn;     // B*G*3
  float* mask    = centers + (size_t)NGRP * 3;   // B*G

  float* W      = (float*)d_ws;
  float* grp    = W;                             // 4096*32*3 = 393216 floats
  float* stats  = W + (size_t)NGRP * Sn * 3;
  float* sums1  = stats + 0;
  float* sumsq1 = stats + 128;
  float* scale1 = stats + 256;
  float* shift1 = stats + 384;
  float* sums3  = stats + 512;
  float* sumsq3 = stats + 1024;
  float* scale3 = stats + 1536;
  float* shift3 = stats + 2048;
  _Float16* w2h = (_Float16*)(stats + 2560);     // 256*128
  _Float16* w3h = w2h + 256 * 128;               // 512*512
  _Float16* w4h = w3h + 512 * 512;               // 384*512

  fps_kernel<<<Bn, 512, 0, stream>>>(points, centers);
  knn_kernel<<<NGRP, 256, 0, stream>>>(points, centers, grp);

  zero_kernel<<<(2560 + 255) / 256, 256, 0, stream>>>(stats, 2560);
  f32_to_f16_kernel<<<(256 * 128 + 255) / 256, 256, 0, stream>>>(w2, w2h, 256 * 128);
  f32_to_f16_kernel<<<(512 * 512 + 255) / 256, 256, 0, stream>>>(w3, w3h, 512 * 512);
  f32_to_f16_kernel<<<(384 * 512 + 255) / 256, 256, 0, stream>>>(w4, w4h, 384 * 512);

  stats1_kernel<<<NGRP, 128, 0, stream>>>(grp, w1, sums1, sumsq1);
  bn_finalize_kernel<<<1, 128, 0, stream>>>(sums1, sumsq1, g1, be1, scale1, shift1, 128);

  pointnet_kernel<0><<<NGRP, 256, 0, stream>>>(grp, w1, scale1, shift1, w2h, b2,
                                               w3h, sums3, sumsq3, scale3, shift3,
                                               w4h, b4, tokens, mask);
  bn_finalize_kernel<<<2, 256, 0, stream>>>(sums3, sumsq3, g3, be3, scale3, shift3, 512);

  pointnet_kernel<1><<<NGRP, 256, 0, stream>>>(grp, w1, scale1, shift1, w2h, b2,
                                               w3h, sums3, sumsq3, scale3, shift3,
                                               w4h, b4, tokens, mask);
}
